// MPSClassifier_77412490543199
// MI455X (gfx1250) — matrix operand order repaired and checked by reading, not compiled, not run on hardware
//
#include <hip/hip_runtime.h>
#include <hip/hip_bf16.h>

// Shapes from reference
#define BATCH   16
#define NPIX    784
#define DBOND   128
#define NSITES  783          // matrices for sites 1..783
#define LGRP    16
#define NGRP    49           // ceil(783/16)
#define NCLS    10

#define SP  65               // P row stride (dwords) = 128 bf16 + pad
#define SPE 130              // P row stride (bf16 elements)
#define SM  65               // Mt (column-major M) row stride (dwords)

typedef __attribute__((ext_vector_type(16))) __bf16 v16bf;
typedef __attribute__((ext_vector_type(8)))  float  v8f;

union Frag { unsigned int u[8]; v16bf v; };

// ---------------------------------------------------------------------------
// Stage 1: one block per (group g, batch b). Chain P = I * M_{g*16} * ... in
// bf16 WMMA with f32 accumulation; write 128x128 bf16 group product to ws.
// ---------------------------------------------------------------------------
__global__ __launch_bounds__(256)
void mps_stage1(const float* __restrict__ x,           // (16, 784)
                const float* __restrict__ cores_rest,  // (783, 128, 2, 128)
                unsigned short* __restrict__ gprod)    // (NGRP*BATCH, 128*128) bf16 bits
{
    __shared__ unsigned int Pbuf[2][DBOND * SP];   // product, row-major, bf16 pairs
    __shared__ unsigned int Mt[DBOND * SM];        // M transposed (row = n), bf16 pairs

    const int bid  = blockIdx.x;
    const int g    = bid / BATCH;
    const int b    = bid % BATCH;
    const int tid  = threadIdx.x;
    const int lane = tid & 31;
    const int wave = tid >> 5;                     // 0..7

    // ---- init P = identity (bf16) ----
    __bf16* Pe0 = (__bf16*)&Pbuf[0][0];
    for (int i = tid; i < DBOND * SPE; i += 256) Pe0[i] = (__bf16)0.0f;
    if (tid < DBOND) Pe0[tid * SPE + tid] = (__bf16)1.0f;

    int cur = 0;
    const int nsite = min(LGRP, NSITES - g * LGRP);

    // fragment addressing constants (contiguous k: lane half h holds k 16h..16h+15 of each chunk)
    const int rA     = wave * 16 + (lane & 15);        // A row
    const int koffA  = (lane < 16) ? 0 : 8;            // dword offset within K-chunk
    const int nB     = (lane & 15);                    // B column (within tile)
    const int kbaseB = (lane < 16) ? 0 : 8;            // dword offset within K-chunk
    const int cnD    = (lane & 15);                    // D column
    const int roffD  = wave * 16 + ((lane < 16) ? 0 : 8);

    for (int s = 0; s < nsite; ++s) {
        const int j = g * LGRP + s;                    // matrix index 0..782
        const float p  = x[b * NPIX + (j + 1)];
        const float f0 = 1.0f - p;
        const float f1 = p;
        const float* Aj = cores_rest + (size_t)j * (DBOND * 2 * DBOND);

        // ---- build Mt (column-major M, bf16 K-pairs packed per dword) ----
        // task t: kp = t>>5 (row pair 2kp,2kp+1), nblk = t&31 (cols 4*nblk..+3)
        #pragma unroll
        for (int it = 0; it < 8; ++it) {
            const int t    = it * 256 + tid;
            const int nblk = t & 31;
            const int kp   = t >> 5;
            const int n0   = nblk * 4;
            const float4 a0 = *(const float4*)(Aj + (size_t)(2 * kp)     * 256 + n0);
            const float4 b0 = *(const float4*)(Aj + (size_t)(2 * kp)     * 256 + 128 + n0);
            const float4 a1 = *(const float4*)(Aj + (size_t)(2 * kp + 1) * 256 + n0);
            const float4 b1 = *(const float4*)(Aj + (size_t)(2 * kp + 1) * 256 + 128 + n0);
            union { unsigned int u; __bf16 h[2]; } pk;
            pk.h[0] = (__bf16)(f0 * a0.x + f1 * b0.x);
            pk.h[1] = (__bf16)(f0 * a1.x + f1 * b1.x);
            Mt[(n0 + 0) * SM + kp] = pk.u;
            pk.h[0] = (__bf16)(f0 * a0.y + f1 * b0.y);
            pk.h[1] = (__bf16)(f0 * a1.y + f1 * b1.y);
            Mt[(n0 + 1) * SM + kp] = pk.u;
            pk.h[0] = (__bf16)(f0 * a0.z + f1 * b0.z);
            pk.h[1] = (__bf16)(f0 * a1.z + f1 * b1.z);
            Mt[(n0 + 2) * SM + kp] = pk.u;
            pk.h[0] = (__bf16)(f0 * a0.w + f1 * b0.w);
            pk.h[1] = (__bf16)(f0 * a1.w + f1 * b1.w);
            Mt[(n0 + 3) * SM + kp] = pk.u;
        }
        __syncthreads();

        // ---- P_next = P_cur @ M via v_wmma_f32_16x16x32_bf16 ----
        const unsigned int* P = &Pbuf[cur][0];
        v8f acc[8] = {};
        #pragma unroll
        for (int kc = 0; kc < 4; ++kc) {
            Frag afr;
            const unsigned int* pb = P + rA * SP + kc * 16 + koffA;
            #pragma unroll
            for (int i = 0; i < 4; ++i) afr.u[i]     = pb[i];
            #pragma unroll
            for (int i = 0; i < 4; ++i) afr.u[4 + i] = pb[4 + i];
            #pragma unroll
            for (int t = 0; t < 8; ++t) {
                Frag bfr;
                const unsigned int* mb = &Mt[(t * 16 + nB) * SM + kc * 16 + kbaseB];
                #pragma unroll
                for (int i = 0; i < 8; ++i) bfr.u[i] = mb[i];
                acc[t] = __builtin_amdgcn_wmma_f32_16x16x32_bf16(
                    false, afr.v, false, bfr.v, (short)0, acc[t], false, false);
            }
        }

        // ---- store D (f32) -> P_next (bf16, row-major) ----
        __bf16* Pn = (__bf16*)&Pbuf[cur ^ 1][0];
        #pragma unroll
        for (int t = 0; t < 8; ++t) {
            #pragma unroll
            for (int i = 0; i < 8; ++i)
                Pn[(roffD + i) * SPE + t * 16 + cnD] = (__bf16)acc[t][i];
        }
        __syncthreads();
        cur ^= 1;
    }

    // ---- write group product (unpadded, row-major bf16) ----
    unsigned int* out = (unsigned int*)(gprod + (size_t)bid * (DBOND * DBOND));
    const unsigned int* Psrc = &Pbuf[cur][0];
    for (int dd = tid; dd < DBOND * 64; dd += 256) {       // 64 dwords per row
        const int r = dd >> 6;
        const int c = dd & 63;
        out[dd] = Psrc[r * SP + c];
    }
}

// ---------------------------------------------------------------------------
// Stage 2: one block per batch. v = eff0; v = v @ P_g (49 groups); logits.
// ---------------------------------------------------------------------------
__global__ __launch_bounds__(128)
void mps_stage2(const float* __restrict__ x,
                const float* __restrict__ core_first,   // (1, 2, 128)
                const float* __restrict__ classifier,   // (10, 128)
                const unsigned short* __restrict__ gprod,
                float* __restrict__ out)                // (16, 10)
{
    __shared__ float v[2][DBOND];
    const int b   = blockIdx.x;
    const int tid = threadIdx.x;                        // 0..127

    const float p0 = x[b * NPIX + 0];
    v[0][tid] = (1.0f - p0) * core_first[tid] + p0 * core_first[DBOND + tid];
    __syncthreads();

    int cur = 0;
    for (int g = 0; g < NGRP; ++g) {
        const unsigned short* P = gprod + (size_t)(g * BATCH + b) * (DBOND * DBOND);
        float acc = 0.0f;
        #pragma unroll 4
        for (int k = 0; k < DBOND; ++k) {
            const unsigned int u = ((unsigned int)P[k * DBOND + tid]) << 16;
            acc += v[cur][k] * __uint_as_float(u);
        }
        v[cur ^ 1][tid] = acc;
        __syncthreads();
        cur ^= 1;
    }

    if (tid < NCLS) {
        float acc = 0.0f;
        #pragma unroll 4
        for (int k = 0; k < DBOND; ++k)
            acc += v[cur][k] * classifier[tid * DBOND + k];
        out[b * NCLS + tid] = acc;
    }
}

// ---------------------------------------------------------------------------
extern "C" void kernel_launch(void* const* d_in, const int* in_sizes, int n_in,
                              void* d_out, int out_size, void* d_ws, size_t ws_size,
                              hipStream_t stream) {
    const float* x          = (const float*)d_in[0];   // (16, 784)
    const float* core_first = (const float*)d_in[1];   // (1, 2, 128)
    const float* cores_rest = (const float*)d_in[2];   // (783, 128, 2, 128)
    const float* classifier = (const float*)d_in[3];   // (10, 128)
    unsigned short* gprod   = (unsigned short*)d_ws;   // 784 * 128*128 bf16 = 25.7 MB

    mps_stage1<<<NGRP * BATCH, 256, 0, stream>>>(x, cores_rest, gprod);
    mps_stage2<<<BATCH, 128, 0, stream>>>(x, core_first, classifier, gprod,
                                          (float*)d_out);
}
